// CustomRNN_29489245454593
// MI455X (gfx1250) — matrix operand order repaired and checked by reading, not compiled, not run on hardware
//
#include <hip/hip_runtime.h>
#include <hip/hip_bf16.h>

// ---------------------------------------------------------------------------
// Linear-RNN scan for MI455X (gfx1250, wave32, WMMA).
//   h_{t+1} = h_t * W2^T + x_t (x) w1 ;  y_t = h_{t+1} * W3^T
// Affine-scan chunking: 128 parallel chunks of 64 steps, one serial combine
// with M^64 (6 WMMA squarings), then a parallel correction pass.
// All GEMMs: v_wmma_f32_16x16x32_f16, operands in LDS, layout-exact loads.
// New this round: 4-deep h ring + batched y projection (uniform 136 WMMA per
// wave per 4 steps, no straggler waves) and prefetch on the serial combine.
// ---------------------------------------------------------------------------

typedef __attribute__((ext_vector_type(16))) _Float16 v16h;
typedef __attribute__((ext_vector_type(8)))  _Float16 v8h;
typedef __attribute__((ext_vector_type(8)))  float    v8f;

#define BB        64      // batch
#define TT        8192    // time steps
#define HH        256     // hidden
#define OUTD      10      // output dim
#define CHUNK     64      // timesteps per chunk (optimum: sqrt(T/2))
#define NCHUNK    (TT / CHUNK)   // 128 chunks
#define NTHREADS  512            // 16 wave32s
#define KTILES    (HH / 32)      // 8 K-steps of 32 per 16x16 tile

__device__ __forceinline__ v8f wmma_f16(v16h a, v16h b, v8f c) {
  return __builtin_amdgcn_wmma_f32_16x16x32_f16(
      /*neg_a=*/false, a, /*neg_b=*/false, b,
      /*c_mod=*/(short)0, c, /*reuse_a=*/false, /*reuse_b=*/false);
}

// A-operand (16x32 f16) fragment from a row-major [rows][HH] f16 LDS buffer.
// Layout used (contiguous k, same in A and B): M = lane&15 (+16*mi), K = kt*32 + 16*hi + e -> contiguous 32B.
__device__ __forceinline__ v16h load_a16(const _Float16* __restrict__ base,
                                         int mi, int kt, int lane) {
  const int hi = lane >> 4;
  const int m  = (lane & 15) + (mi << 4);
  const _Float16* row = base + m * HH + kt * 32 + (hi << 4);
  return *(const v16h*)(row);
}

// B-operand (32x16 f16) for  X @ Y^T  with Y row-major [N][HH] in LDS.
// Layout used (contiguous k, same in A and B): N = lane&15 (+16*ni), K = kt*32 + 16*hi + e -> contiguous 32B.
__device__ __forceinline__ v16h load_bT16(const _Float16* __restrict__ Y,
                                          int ni, int kt, int lane) {
  const int hi = lane >> 4;
  const int n  = (lane & 15) + (ni << 4);
  return *(const v16h*)(Y + n * HH + kt * 32 + (hi << 4));
}

// ---------------------------------------------------------------------------
// Phase 1 / Phase 3 kernel. One block per chunk, 16 waves.
// Recurrence: wave w owns tiles (mi = w&3, ni = (w>>2)+4q).
// y projection: batched per 4 steps over the h ring; wave w owns
// (step-in-group = w>>2, batch-tile = w&3).
// ---------------------------------------------------------------------------
template <bool PHASE1>
__global__ __launch_bounds__(NTHREADS)
void rnn_chunk_kernel(const float* __restrict__ x,
                      const float* __restrict__ W1,
                      const float* __restrict__ W2,
                      const float* __restrict__ W3,
                      float* __restrict__ out,
                      float* __restrict__ Lend,            // [NCHUNK][BB][HH] f32
                      const _Float16* __restrict__ hstart) // [NCHUNK][BB][HH] f16
{
  __shared__ _Float16 sW2[HH * HH];        // 128 KB
  __shared__ _Float16 sW3[16 * HH];        //   8 KB (padded to 16 rows)
  __shared__ _Float16 sHr[4][BB * HH];     // 128 KB : 4-deep ring of h (f16)
  __shared__ float    sX[CHUNK * BB];      //  16 KB

  const int tid  = threadIdx.x;
  const int lane = tid & 31;
  const int wave = tid >> 5;
  const int c    = blockIdx.x;
  const int t0   = c * CHUNK;

  for (int i = tid; i < HH * HH; i += NTHREADS) sW2[i] = (_Float16)W2[i];
  for (int i = tid; i < 16 * HH; i += NTHREADS) {
    const int o = i >> 8, k = i & (HH - 1);
    sW3[i] = (o < OUTD) ? (_Float16)W3[o * HH + k] : (_Float16)0.f;
  }
  if (PHASE1) {
    for (int i = tid; i < CHUNK * BB; i += NTHREADS) {
      const int s = i >> 6, b = i & 63;
      sX[i] = x[(size_t)b * TT + t0 + s];
    }
    for (int i = tid; i < BB * HH; i += NTHREADS) sHr[3][i] = (_Float16)0.f;
  } else {
    for (int i = tid; i < BB * HH; i += NTHREADS)
      sHr[3][i] = hstart[(size_t)c * (BB * HH) + i];
  }

  const int mi = wave & 3;   // recurrence batch tile
  const int nb = wave >> 2;  // recurrence hidden tile base
  const int ysl = wave >> 2; // y: step-in-group
  const int ybt = wave & 3;  // y: batch tile

  float w1v[4] = {0.f, 0.f, 0.f, 0.f};
  if (PHASE1) {
#pragma unroll
    for (int q = 0; q < 4; ++q) w1v[q] = W1[((nb + 4 * q) << 4) + (lane & 15)];
  }

  __syncthreads();

  const int hi = lane >> 4;
  const int ln = lane & 15;

  for (int sb = 0; sb < CHUNK; sb += 4) {
#pragma unroll
    for (int j = 0; j < 4; ++j) {          // step s = sb + j, ring slot j
      const _Float16* hprev = sHr[(j + 3) & 3];

      v8f acc[4];
#pragma unroll
      for (int q = 0; q < 4; ++q) acc[q] = v8f{};

#pragma unroll
      for (int kt = 0; kt < KTILES; ++kt) {
        const v16h a = load_a16(hprev, mi, kt, lane);
#pragma unroll
        for (int q = 0; q < 4; ++q) {
          const v16h b = load_bT16(sW2, nb + 4 * q, kt, lane);
          acc[q] = wmma_f16(a, b, acc[q]);
        }
      }

      if (PHASE1) {  // += x_t (outer) w1
        const int s = sb + j;
#pragma unroll
        for (int q = 0; q < 4; ++q)
#pragma unroll
          for (int r = 0; r < 8; ++r)
            acc[q][r] += sX[s * BB + (mi << 4) + (hi << 3) + r] * w1v[q];
      }

      __syncthreads();  // prior readers of slot j are done

#pragma unroll
      for (int q = 0; q < 4; ++q) {
        const int n = ((nb + 4 * q) << 4) + ln;
#pragma unroll
        for (int r = 0; r < 8; ++r) {
          const int m = (mi << 4) + (hi << 3) + r;
          sHr[j][m * HH + n] = (_Float16)acc[q][r];
        }
      }
      if (PHASE1 && j == 3 && sb == CHUNK - 4) {  // chunk-end local state
#pragma unroll
        for (int q = 0; q < 4; ++q) {
          const int n = ((nb + 4 * q) << 4) + ln;
#pragma unroll
          for (int r = 0; r < 8; ++r) {
            const int m = (mi << 4) + (hi << 3) + r;
            Lend[((size_t)c * BB + m) * HH + n] = acc[q][r];
          }
        }
      }

      __syncthreads();  // slot j fully written
    }

    // Batched y projection over the 4 fresh h states: uniform across 16 waves.
    {
      v8f ya = v8f{};
#pragma unroll
      for (int kt = 0; kt < KTILES; ++kt) {
        const v16h a = load_a16(sHr[ysl], ybt, kt, lane);
        const v16h b = load_bT16(sW3, 0, kt, lane);
        ya = wmma_f16(a, b, ya);
      }
      if (ln < OUTD) {
        const int t = t0 + sb + ysl;
#pragma unroll
        for (int r = 0; r < 8; ++r) {
          const int bg = (ybt << 4) + (hi << 3) + r;
          const size_t idx = ((size_t)bg * TT + t) * OUTD + ln;
          if (PHASE1) out[idx] = ya[r];
          else        out[idx] += ya[r];  // correction pass
        }
      }
    }
    // Ring-slot 0 is next overwritten after the next step's pre-write barrier,
    // which orders it after these y reads. No extra barrier needed here.
  }
}

// ---------------------------------------------------------------------------
// Phase 2: single block. Square M = W2^T six times -> M^64, then serially
// combine chunk states: hstart[c+1] = hstart[c] * M^64 + Lend[c].
// ---------------------------------------------------------------------------
__global__ __launch_bounds__(NTHREADS)
void rnn_combine_kernel(const float* __restrict__ W2,
                        const float* __restrict__ Lend,
                        _Float16* __restrict__ hstart)
{
  __shared__ _Float16 sP [HH * HH];  // 128 KB : M^(2^j)
  __shared__ _Float16 sPt[HH * HH];  // 128 KB : its transpose
  __shared__ _Float16 sH [BB * HH];  //  32 KB

  const int tid  = threadIdx.x;
  const int lane = tid & 31;
  const int wave = tid >> 5;
  const int hi   = lane >> 4;
  const int ln   = lane & 15;

  for (int i = tid; i < HH * HH; i += NTHREADS) {
    sPt[i] = (_Float16)W2[i];                 // M^T = W2
    const int m = i >> 8, k = i & 255;
    sP[i] = (_Float16)W2[k * HH + m];         // M = W2^T
  }
  __syncthreads();

  // 6 squarings: Pnew = P*P = P @ (P^T)^T (A rows from sP, B rows from sPt)
  for (int it = 0; it < 6; ++it) {
    v8f acc[16];
#pragma unroll
    for (int ni = 0; ni < 16; ++ni) acc[ni] = v8f{};
#pragma unroll
    for (int kt = 0; kt < KTILES; ++kt) {
      const v16h a = load_a16(sP, wave, kt, lane);
#pragma unroll
      for (int ni = 0; ni < 16; ++ni) {
        const v16h b = load_bT16(sPt, ni, kt, lane);
        acc[ni] = wmma_f16(a, b, acc[ni]);
      }
    }
    __syncthreads();
#pragma unroll
    for (int ni = 0; ni < 16; ++ni) {
      const int n = (ni << 4) + ln;
#pragma unroll
      for (int r = 0; r < 8; ++r) {
        const int m = (wave << 4) + (hi << 3) + r;
        const _Float16 v = (_Float16)acc[ni][r];
        sP [m * HH + n] = v;
        sPt[n * HH + m] = v;
      }
    }
    __syncthreads();
  }

  // Serial combine over chunks (the true sequential bottleneck: hide the
  // Lend global-load latency with one-iteration-ahead prefetch).
  const int mi = wave & 3;
  const int nb = wave >> 2;

  for (int i = tid; i < BB * HH; i += NTHREADS) {
    sH[i] = (_Float16)0.f;
    hstart[i] = (_Float16)0.f;  // hstart[0] = 0
  }
  __syncthreads();

  for (int c = 0; c < NCHUNK - 1; ++c) {
    // Prefetch next chunk's Lend block (64 KB = 512 lines, 1 line/thread).
    // Lowers to global_prefetch_b8 on gfx1250.
    if (c + 1 < NCHUNK) {
      const char* nxt = (const char*)(Lend + (size_t)(c + 1) * BB * HH);
      __builtin_prefetch(nxt + (size_t)tid * 128, 0, 3);
    }

    v8f acc[4];
#pragma unroll
    for (int q = 0; q < 4; ++q) acc[q] = v8f{};
#pragma unroll
    for (int kt = 0; kt < KTILES; ++kt) {
      const v16h a = load_a16(sH, mi, kt, lane);
#pragma unroll
      for (int q = 0; q < 4; ++q) {
        const v16h b = load_bT16(sPt, nb + 4 * q, kt, lane);
        acc[q] = wmma_f16(a, b, acc[q]);
      }
    }
#pragma unroll
    for (int q = 0; q < 4; ++q) {
      const int n = ((nb + 4 * q) << 4) + ln;
#pragma unroll
      for (int r = 0; r < 8; ++r) {
        const int m = (mi << 4) + (hi << 3) + r;
        acc[q][r] += Lend[((size_t)c * BB + m) * HH + n];
      }
    }
    __syncthreads();
#pragma unroll
    for (int q = 0; q < 4; ++q) {
      const int n = ((nb + 4 * q) << 4) + ln;
#pragma unroll
      for (int r = 0; r < 8; ++r) {
        const int m = (mi << 4) + (hi << 3) + r;
        const _Float16 v = (_Float16)acc[q][r];
        sH[m * HH + n] = v;
        hstart[((size_t)(c + 1) * BB + m) * HH + n] = v;
      }
    }
    __syncthreads();
  }
}

// ---------------------------------------------------------------------------
// Workspace: [0, 8MB) Lend f32 ; [8MB, 12MB) hstart f16.
// ---------------------------------------------------------------------------
extern "C" void kernel_launch(void* const* d_in, const int* in_sizes, int n_in,
                              void* d_out, int out_size, void* d_ws, size_t ws_size,
                              hipStream_t stream) {
  const float* x  = (const float*)d_in[0];
  const float* W1 = (const float*)d_in[1];
  const float* W2 = (const float*)d_in[2];
  const float* W3 = (const float*)d_in[3];
  float* out = (float*)d_out;

  float*    Lend   = (float*)d_ws;
  _Float16* hstart = (_Float16*)((char*)d_ws + (size_t)NCHUNK * BB * HH * 4);

  rnn_chunk_kernel<true><<<NCHUNK, NTHREADS, 0, stream>>>(
      x, W1, W2, W3, out, Lend, nullptr);
  rnn_combine_kernel<<<1, NTHREADS, 0, stream>>>(W2, Lend, hstart);
  rnn_chunk_kernel<false><<<NCHUNK, NTHREADS, 0, stream>>>(
      nullptr, nullptr, W2, W3, out, nullptr, hstart);
}
